// ContentAgree_1159641170232
// MI455X (gfx1250) — matrix operand order repaired and checked by reading, not compiled, not run on hardware
//
#include <hip/hip_runtime.h>
#include <hip/hip_bf16.h>
#include <math.h>

#define DEVINL __device__ __forceinline__

typedef __attribute__((ext_vector_type(16))) _Float16 v16h;
typedef __attribute__((ext_vector_type(8)))  _Float16 h8;
typedef __attribute__((ext_vector_type(8)))  float    v8f;
typedef __attribute__((ext_vector_type(4)))  unsigned int u32x4;
typedef __attribute__((ext_vector_type(8)))  int      i32x8;
typedef __attribute__((ext_vector_type(4)))  int      i32x4;

#ifndef __has_builtin
#define __has_builtin(x) 0
#endif
#if __has_builtin(__builtin_amdgcn_tensor_load_to_lds)
#define HAVE_TDM 1
#else
#define HAVE_TDM 0
#endif

namespace cfg {
constexpr int B = 4, DIM = 256, N = 16384, K = 64, H = 4, HD = DIM / H, FF = 2 * DIM;
constexpr float SCALE = 0.125f;  // 1/sqrt(HD)
constexpr int NCH = 8;           // split-K chunks for the attn3@V GEMM
}
using namespace cfg;

DEVINL v8f zero8() { v8f z = {0.f,0.f,0.f,0.f,0.f,0.f,0.f,0.f}; return z; }

DEVINL v8f wmma16(v16h a, v16h b, v8f c) {
  return __builtin_amdgcn_wmma_f32_16x16x32_f16(
      false, a, false, b, (short)0, c, false, false);
}

DEVINL v16h cat16(h8 lo, h8 hi) {
  return __builtin_shufflevector(lo, hi, 0,1,2,3,4,5,6,7,8,9,10,11,12,13,14,15);
}

// ---- f16 fragment builders: two 16B vector loads each (ISA 7.12.2 wave32 layouts) ----

// A fragment from row-major f16: A[r][k] = src[r*ld + k]
DEVINL v16h a_frag_h(const _Float16* __restrict__ src, int ld) {
  int lane = threadIdx.x & 31;
  const _Float16* p = src + (lane & 15) * ld + ((lane >> 4) << 3);
  return cat16(*(const h8*)p, *(const h8*)(p + 16));
}

// B fragment from k-contiguous f16 source: B[k][c] = src[c*ld + k]
DEVINL v16h b_frag_hT(const _Float16* __restrict__ src, int ld) {
  int lane = threadIdx.x & 31;
  const _Float16* p = src + (lane & 15) * ld + ((lane >> 4) << 3);
  return cat16(*(const h8*)p, *(const h8*)(p + 16));
}

// ---- f32 gather fragment builders (only for the tiny C@Wq GEMM) ----
DEVINL v16h a_frag_rm(const float* src, int ld) {
  int lane = threadIdx.x & 31, r = lane & 15, kb = (lane >> 4) << 3;
  v16h a;
#pragma unroll
  for (int v = 0; v < 8; ++v) {
    int k = kb + ((v < 4) ? (2 * v) : (16 + 2 * (v - 4)));
    a[2 * v] = (_Float16)src[r * ld + k];
    a[2 * v + 1] = (_Float16)src[r * ld + k + 1];
  }
  return a;
}
DEVINL v16h b_frag_rm(const float* src, int ld) {
  int lane = threadIdx.x & 31, c = lane & 15, kb = (lane >> 4) << 3;
  v16h b;
#pragma unroll
  for (int v = 0; v < 8; ++v) {
    b[2 * v] = (_Float16)src[(kb + 2 * v + ((v < 4) ? 0 : 8)) * ld + c];
    b[2 * v + 1] = (_Float16)src[(kb + 2 * v + 1 + ((v < 4) ? 0 : 8)) * ld + c];
  }
  return b;
}

// ---- Tensor Data Mover: stage a [rows x cols] f16 row-major strip into LDS ----
// D# per ISA 08 §8.3/8.4: group0={count,lds,global,type=2}; group1={data_size=2B,
// tensor_dim0=cols, tensor_dim1=rows, tile_dim0=cols, tile_dim1=rows, stride0=cols};
// groups 2/3 zero (2-D tensor). Issued by one wave (EXEC-ignored), waited with
// s_wait_tensorcnt, then block-barriered.
DEVINL void tdm_stage_rows(_Float16* lds_dst, const _Float16* gsrc, int rows, int cols) {
#if HAVE_TDM
  if (threadIdx.x == 0) {
    unsigned long long ga = (unsigned long long)gsrc;
    unsigned lds = (unsigned)(unsigned long long)lds_dst;
    u32x4 g0;
    g0[0] = 1u;                                    // count=1, is_restore=0
    g0[1] = lds;                                   // lds_addr
    g0[2] = (unsigned)ga;                          // global_addr[31:0]
    g0[3] = (unsigned)((ga >> 32) & 0x1FFFFFFull) | (2u << 30);  // addr[56:32] | type=2
    i32x8 g1;
    g1[0] = 1 << 16;                               // workgroup_mask=0, data_size=1 (2B)
    g1[1] = (cols & 0xFFFF) << 16;                 // tensor_dim0[15:0]
    g1[2] = ((cols >> 16) & 0xFFFF) | ((rows & 0xFFFF) << 16);   // dim0 hi | dim1 lo
    g1[3] = ((rows >> 16) & 0xFFFF) | ((cols & 0xFFFF) << 16);   // dim1 hi | tile_dim0
    g1[4] = rows & 0xFFFF;                         // tile_dim1 | tile_dim2=0
    g1[5] = cols;                                  // tensor_dim0_stride[31:0]
    g1[6] = 0; g1[7] = 0;
    i32x4 gz = {0, 0, 0, 0};
#if __clang_major__ >= 23
    i32x8 gz8 = {0, 0, 0, 0, 0, 0, 0, 0};
    __builtin_amdgcn_tensor_load_to_lds(g0, g1, gz, gz, gz8, 0);
#else
    __builtin_amdgcn_tensor_load_to_lds(g0, g1, gz, gz, 0);
#endif
#if __has_builtin(__builtin_amdgcn_s_wait_tensorcnt)
    __builtin_amdgcn_s_wait_tensorcnt(0);
#endif
  }
#else
  int n8 = rows * cols / 8;
  for (int i = threadIdx.x; i < n8; i += blockDim.x)
    ((h8*)lds_dst)[i] = ((const h8*)gsrc)[i];
#endif
  __syncthreads();
}

// ================= transposes (32x32 LDS tiles, coalesced both sides) =================

__global__ void k_tr_f2h(const float* __restrict__ src, _Float16* __restrict__ dst,
                         int R, int C) {   // dst[c*R + r] = (f16)src[r*C + c], batched on z
  __shared__ _Float16 t[32][33];
  long sb = (long)blockIdx.z * R * C;
  int c0 = blockIdx.x * 32, r0 = blockIdx.y * 32;
  int tx = threadIdx.x, ty = threadIdx.y;
#pragma unroll
  for (int i = 0; i < 32; i += 8)
    t[ty + i][tx] = (_Float16)src[sb + (long)(r0 + ty + i) * C + (c0 + tx)];
  __syncthreads();
#pragma unroll
  for (int i = 0; i < 32; i += 8)
    dst[sb + (long)(c0 + ty + i) * R + (r0 + tx)] = t[tx][ty + i];
}

__global__ void k_tr_h2h(const _Float16* __restrict__ src, _Float16* __restrict__ dst,
                         int R, int C) {
  __shared__ _Float16 t[32][33];
  long sb = (long)blockIdx.z * R * C;
  int c0 = blockIdx.x * 32, r0 = blockIdx.y * 32;
  int tx = threadIdx.x, ty = threadIdx.y;
#pragma unroll
  for (int i = 0; i < 32; i += 8)
    t[ty + i][tx] = src[sb + (long)(r0 + ty + i) * C + (c0 + tx)];
  __syncthreads();
#pragma unroll
  for (int i = 0; i < 32; i += 8)
    dst[sb + (long)(c0 + ty + i) * R + (r0 + tx)] = t[tx][ty + i];
}

// ================= tiny GEMMs =================

__global__ void k_proj_k1(const float* __restrict__ Cp, const float* __restrict__ Wq,
                          const float* __restrict__ bq, _Float16* __restrict__ K1h) {
  int t = blockIdx.x;
  int j = t & 15; t >>= 4;
  int ki = t & 3; t >>= 2;
  int b = t;
  const float* Ab = Cp + b * K * DIM + ki * 16 * DIM;
  v8f acc = zero8();
  for (int kk = 0; kk < DIM; kk += 32)
    acc = wmma16(a_frag_rm(Ab + kk, DIM), b_frag_rm(Wq + kk * DIM + j * 16, DIM), acc);
  int lane = threadIdx.x & 31, col = lane & 15, rb = (lane >> 4) * 8;
  float bias = bq[j * 16 + col];
  _Float16* dst = K1h + b * K * DIM + ki * 16 * DIM + j * 16;
#pragma unroll
  for (int v = 0; v < 8; ++v) dst[(rb + v) * DIM + col] = (_Float16)(acc[v] + bias);
}

__global__ void k_attn2(const _Float16* __restrict__ K1h, _Float16* __restrict__ attn2) {
  __shared__ float S[K * K];
  int bh = blockIdx.x, b = bh / H, h = bh % H;
  const _Float16* Kh = K1h + b * K * DIM + h * HD;
  int w = threadIdx.x >> 5, ti = w >> 2, tj = w & 3;
  v8f acc = zero8();
#pragma unroll
  for (int kk = 0; kk < HD; kk += 32)
    acc = wmma16(a_frag_h(Kh + ti * 16 * DIM + kk, DIM),
                 b_frag_hT(Kh + tj * 16 * DIM + kk, DIM), acc);
  int lane = threadIdx.x & 31, col = lane & 15, rb = (lane >> 4) * 8;
#pragma unroll
  for (int v = 0; v < 8; ++v) S[(ti * 16 + rb + v) * K + tj * 16 + col] = acc[v] * SCALE;
  __syncthreads();
  if (threadIdx.x < K) {
    float* row = S + threadIdx.x * K;
    float m = row[0];
    for (int i = 1; i < K; ++i) m = fmaxf(m, row[i]);
    float s = 0.f;
    for (int i = 0; i < K; ++i) { float e = __expf(row[i] - m); row[i] = e; s += e; }
    float inv = 1.0f / s;
    _Float16* dst = attn2 + bh * K * K + threadIdx.x * K;
    for (int i = 0; i < K; ++i) dst[i] = (_Float16)(row[i] * inv);
  }
}

// ================= projections: 16x64 per wave (4 accumulators) =================

__global__ void k_qv(const _Float16* __restrict__ xs16, const _Float16* __restrict__ WT,
                     const float* __restrict__ bias, _Float16* __restrict__ D16) {
  int wid = blockIdx.x * 8 + (threadIdx.x >> 5);
  int jg = wid & 3; wid >>= 2;
  int nt = wid & 1023, b = wid >> 10;
  const _Float16* Ab = xs16 + ((long)(b * N) + nt * 16) * DIM;
  v8f acc[4] = {zero8(), zero8(), zero8(), zero8()};
  for (int kk = 0; kk < DIM; kk += 32) {
    __builtin_prefetch(Ab + kk + 64, 0, 1);
    v16h a = a_frag_h(Ab + kk, DIM);
#pragma unroll
    for (int j = 0; j < 4; ++j)
      acc[j] = wmma16(a, b_frag_hT(WT + (long)(jg * 64 + j * 16) * DIM + kk, DIM), acc[j]);
  }
  int lane = threadIdx.x & 31, col = lane & 15, rb = (lane >> 4) * 8;
#pragma unroll
  for (int j = 0; j < 4; ++j) {
    float bv = bias[jg * 64 + j * 16 + col];
#pragma unroll
    for (int v = 0; v < 8; ++v)
      D16[((long)(b * N) + nt * 16 + rb + v) * DIM + jg * 64 + j * 16 + col] =
          (_Float16)(acc[j][v] + bv);
  }
}

__global__ void k_scores(const _Float16* __restrict__ Q16, const _Float16* __restrict__ K1h,
                         float* __restrict__ Sc) {
  int wid = blockIdx.x * 8 + (threadIdx.x >> 5);
  int nt = wid & 1023, bh = wid >> 10;
  int b = bh / H, h = bh % H;
  const _Float16* Qb = Q16 + ((long)(b * N) + nt * 16) * DIM + h * HD;
  const _Float16* Kb = K1h + (long)b * K * DIM + h * HD;
  v8f acc[4] = {zero8(), zero8(), zero8(), zero8()};
#pragma unroll
  for (int kk = 0; kk < HD; kk += 32) {
    v16h a = a_frag_h(Qb + kk, DIM);
#pragma unroll
    for (int j = 0; j < 4; ++j)
      acc[j] = wmma16(a, b_frag_hT(Kb + (long)(j * 16) * DIM + kk, DIM), acc[j]);
  }
  int lane = threadIdx.x & 31, col = lane & 15, rb = (lane >> 4) * 8;
#pragma unroll
  for (int j = 0; j < 4; ++j)
#pragma unroll
    for (int v = 0; v < 8; ++v)
      Sc[((long)bh * N + nt * 16 + rb + v) * K + j * 16 + col] = acc[j][v] * SCALE;
}

// ================= softmax statistics =================

__global__ void k_rowstats(const float* __restrict__ Sc, float* __restrict__ rmax,
                           float* __restrict__ rsum) {
  long row = (long)blockIdx.x * 256 + threadIdx.x;
  const float* r = Sc + row * K;
  float m = -1e30f;
#pragma unroll
  for (int i = 0; i < K; ++i) m = fmaxf(m, r[i]);
  float s = 0.f;
#pragma unroll
  for (int i = 0; i < K; ++i) s += __expf(r[i] - m);
  rmax[row] = m; rsum[row] = s;
}

__global__ void k_colstats(const float* __restrict__ Sc, float* __restrict__ cmax,
                           float* __restrict__ csum) {
  __shared__ float red[4][64];
  int bh = blockIdx.x;
  int k = threadIdx.x & 63, g = threadIdx.x >> 6;
  const float* base = Sc + (long)bh * N * K;
  float m = -1e30f;
  for (int n = g; n < N; n += 4) m = fmaxf(m, base[(long)n * K + k]);
  red[g][k] = m; __syncthreads();
  if (threadIdx.x < 64) {
    m = fmaxf(fmaxf(red[0][k], red[1][k]), fmaxf(red[2][k], red[3][k]));
    cmax[bh * K + k] = m;
    red[0][k] = m;
  }
  __syncthreads();
  m = red[0][k];
  float s = 0.f;
  for (int n = g; n < N; n += 4) s += __expf(base[(long)n * K + k] - m);
  __syncthreads();
  red[g][k] = s; __syncthreads();
  if (threadIdx.x < 64)
    csum[bh * K + k] = red[0][k] + red[1][k] + red[2][k] + red[3][k];
}

__global__ void k_attn3t(const float* __restrict__ Sc, const float* __restrict__ cmax,
                         _Float16* __restrict__ att3t) {
  __shared__ __align__(16) _Float16 T[256][K];
  __shared__ float cm[K];
  int tile = blockIdx.x;
  int nc = tile & 63, bh = tile >> 6;
  int n0 = nc * 256, t = threadIdx.x;
  if (t < K) cm[t] = cmax[bh * K + t];
  __syncthreads();
  const float* row = Sc + ((long)bh * N + n0 + t) * K;
#pragma unroll 4
  for (int k = 0; k < K; ++k) T[t][k] = (_Float16)__expf(row[k] - cm[k]);
  __syncthreads();
  int k = t & 63, ch = t >> 6;
  _Float16* dst = att3t + ((long)bh * K + k) * N + n0 + ch * 64;
#pragma unroll
  for (int v = 0; v < 8; ++v) {
    h8 tmp;
#pragma unroll
    for (int j = 0; j < 8; ++j) tmp[j] = T[ch * 64 + v * 8 + j][k];
    *(h8*)(dst + v * 8) = tmp;
  }
}

// ================= w2 = attn3 @ V (split-K over N) =================

__global__ void k_w2p(const _Float16* __restrict__ att3t, const _Float16* __restrict__ V1t,
                      float* __restrict__ wpart) {
  int g = blockIdx.x;
  int ch = g & (NCH - 1), bh = g >> 3;
  int b = bh / H, h = bh % H;
  int w = threadIdx.x >> 5;
  const _Float16* ab = att3t + ((long)bh * K + w * 16) * N;
  const _Float16* vb = V1t + ((long)(b * DIM) + h * HD) * N;
  v8f acc[4] = {zero8(), zero8(), zero8(), zero8()};
  int n0 = ch * (N / NCH), n1 = n0 + N / NCH;
  for (int n = n0; n < n1; n += 32) {
    __builtin_prefetch(ab + n + 64, 0, 1);
    v16h a = a_frag_h(ab + n, N);
#pragma unroll
    for (int j = 0; j < 4; ++j)
      acc[j] = wmma16(a, b_frag_hT(vb + (long)(j * 16) * N + n, N), acc[j]);
  }
  int lane = threadIdx.x & 31, col = lane & 15, rb = (lane >> 4) * 8;
#pragma unroll
  for (int j = 0; j < 4; ++j)
#pragma unroll
    for (int v = 0; v < 8; ++v)
      wpart[(((long)ch * 16 + bh) * K + w * 16 + rb + v) * HD + j * 16 + col] = acc[j][v];
}

__global__ void k_w2red(const float* __restrict__ wpart, const float* __restrict__ csum,
                        _Float16* __restrict__ w2T) {
  int idx = blockIdx.x * 256 + threadIdx.x;
  int d = idx & 63, k = (idx >> 6) & 63, bh = idx >> 12;
  float s = 0.f;
#pragma unroll
  for (int ch = 0; ch < NCH; ++ch) s += wpart[(((long)ch * 16 + bh) * K + k) * HD + d];
  w2T[((long)bh * HD + d) * K + k] = (_Float16)(s / csum[bh * K + k]);
}

__global__ void k_matM(const _Float16* __restrict__ attn2, const _Float16* __restrict__ w2T,
                       _Float16* __restrict__ MmT) {
  int bh = blockIdx.x;
  int w = threadIdx.x >> 5, ti = w >> 2, tj = w & 3;
  v8f acc = zero8();
#pragma unroll
  for (int kk = 0; kk < K; kk += 32)
    acc = wmma16(a_frag_h(attn2 + bh * K * K + ti * 16 * K + kk, K),
                 b_frag_hT(w2T + ((long)bh * HD + tj * 16) * K + kk, K), acc);
  int lane = threadIdx.x & 31, col = lane & 15, rb = (lane >> 4) * 8;
#pragma unroll
  for (int v = 0; v < 8; ++v)
    MmT[((long)bh * HD + tj * 16 + col) * K + ti * 16 + rb + v] = (_Float16)acc[v];
}

__global__ void k_att_out(const float* __restrict__ Sc, const float* __restrict__ rmax,
                          const float* __restrict__ rsum, const _Float16* __restrict__ MmT,
                          float* __restrict__ att) {
  int wid = blockIdx.x * 8 + (threadIdx.x >> 5);
  int nt = wid & 1023, bh = wid >> 10;
  int b = bh / H, h = bh % H;
  int lane = threadIdx.x & 31, r = lane & 15, kbA = (lane >> 4) * 8;
  long rowid = (long)bh * N + nt * 16 + r;
  const float* srow = Sc + rowid * K;
  float m = rmax[rowid], inv = 1.0f / rsum[rowid];
  v8f acc[4] = {zero8(), zero8(), zero8(), zero8()};
#pragma unroll
  for (int kk = 0; kk < K; kk += 32) {
    v16h a;
#pragma unroll
    for (int v = 0; v < 8; ++v) {
      int kq = kk + kbA + ((v < 4) ? (2 * v) : (16 + 2 * (v - 4)));
      a[2 * v]     = (_Float16)(__expf(srow[kq] - m) * inv);
      a[2 * v + 1] = (_Float16)(__expf(srow[kq + 1] - m) * inv);
    }
#pragma unroll
    for (int j = 0; j < 4; ++j)
      acc[j] = wmma16(a, b_frag_hT(MmT + ((long)bh * HD + j * 16) * K + kk, K), acc[j]);
  }
  int col = lane & 15, rb = (lane >> 4) * 8;
#pragma unroll
  for (int j = 0; j < 4; ++j)
#pragma unroll
    for (int v = 0; v < 8; ++v)
      att[((long)(b * N) + nt * 16 + rb + v) * DIM + (j * 16 + col) * H + h] = acc[j][v];
}

// ================= FFN =================

__global__ void k_ln1(const float* __restrict__ att, const _Float16* __restrict__ xs16,
                      const float* __restrict__ g, const float* __restrict__ be,
                      float* __restrict__ fin32, _Float16* __restrict__ fin16) {
  __shared__ float red[256];
  long row = blockIdx.x;
  int d = threadIdx.x;
  float val = att[row * DIM + d] + (float)xs16[row * DIM + d];
  red[d] = val; __syncthreads();
  for (int s = 128; s > 0; s >>= 1) { if (d < s) red[d] += red[d + s]; __syncthreads(); }
  float mean = red[0] / DIM; __syncthreads();
  float dv = val - mean;
  red[d] = dv * dv; __syncthreads();
  for (int s = 128; s > 0; s >>= 1) { if (d < s) red[d] += red[d + s]; __syncthreads(); }
  float o = dv * rsqrtf(red[0] / DIM + 1e-5f) * g[d] + be[d];
  fin32[row * DIM + d] = o;
  fin16[row * DIM + d] = (_Float16)o;
}

// hb16 = gelu(fin @ W1 + b1); A-strip (16xDIM) staged to LDS via TDM, shared by 8 waves
__global__ void k_ffn1(const _Float16* __restrict__ fin16, const _Float16* __restrict__ W1T,
                       const float* __restrict__ b1, _Float16* __restrict__ hb16) {
  __shared__ __align__(16) _Float16 Atile[16 * DIM];
  int m0 = blockIdx.x * 16;
  tdm_stage_rows(Atile, fin16 + (long)m0 * DIM, 16, DIM);
  int w = threadIdx.x >> 5;                        // wave = col group of 64 (8 groups)
  v8f acc[4] = {zero8(), zero8(), zero8(), zero8()};
  for (int kk = 0; kk < DIM; kk += 32) {
    v16h a = a_frag_h(Atile + kk, DIM);
#pragma unroll
    for (int j = 0; j < 4; ++j)
      acc[j] = wmma16(a, b_frag_hT(W1T + (long)(w * 64 + j * 16) * DIM + kk, DIM), acc[j]);
  }
  int lane = threadIdx.x & 31, col = lane & 15, rb = (lane >> 4) * 8;
#pragma unroll
  for (int j = 0; j < 4; ++j) {
    float bias = b1[w * 64 + j * 16 + col];
#pragma unroll
    for (int v = 0; v < 8; ++v) {
      float t = acc[j][v] + bias;
      hb16[(long)(m0 + rb + v) * FF + w * 64 + j * 16 + col] =
          (_Float16)(0.5f * t * (1.0f + erff(t * 0.70710678f)));
    }
  }
}

// y = LN(fin + hb @ W2 + b2) -> (B,DIM,N). A-strip (16xFF) staged to LDS via TDM.
__global__ void k_ffn2(const _Float16* __restrict__ hb16, const _Float16* __restrict__ W2T,
                       const float* __restrict__ b2, const float* __restrict__ fin32,
                       const float* __restrict__ g, const float* __restrict__ be,
                       float* __restrict__ out) {
  __shared__ __align__(16) _Float16 Atile[16 * FF];
  __shared__ float S[16 * DIM];
  __shared__ float mu[16], rs[16];
  int m0 = blockIdx.x * 16;
  tdm_stage_rows(Atile, hb16 + (long)m0 * FF, 16, FF);
  int w = threadIdx.x >> 5;                        // 4 col groups of 64
  v8f acc[4] = {zero8(), zero8(), zero8(), zero8()};
  for (int kk = 0; kk < FF; kk += 32) {
    v16h a = a_frag_h(Atile + kk, FF);
#pragma unroll
    for (int j = 0; j < 4; ++j)
      acc[j] = wmma16(a, b_frag_hT(W2T + (long)(w * 64 + j * 16) * FF + kk, FF), acc[j]);
  }
  int lane = threadIdx.x & 31, col = lane & 15, rb = (lane >> 4) * 8;
#pragma unroll
  for (int j = 0; j < 4; ++j) {
    int c = w * 64 + j * 16 + col;
    float bias = b2[c];
#pragma unroll
    for (int v = 0; v < 8; ++v) {
      int r = rb + v;
      S[r * DIM + c] = acc[j][v] + bias + fin32[(long)(m0 + r) * DIM + c];
    }
  }
  __syncthreads();
  if (threadIdx.x < 16) {
    float* row = S + threadIdx.x * DIM;
    float s = 0.f;
    for (int i = 0; i < DIM; ++i) s += row[i];
    float mean = s / DIM;
    float vv = 0.f;
    for (int i = 0; i < DIM; ++i) { float d = row[i] - mean; vv += d * d; }
    mu[threadIdx.x] = mean;
    rs[threadIdx.x] = rsqrtf(vv / DIM + 1e-5f);
  }
  __syncthreads();
  for (int i = threadIdx.x; i < 16 * DIM; i += 128) {
    int r = i >> 8, d = i & 255;
    int grow = m0 + r, b = grow / N, n = grow % N;
    out[((long)(b * DIM) + d) * N + n] = (S[r * DIM + d] - mu[r]) * rs[r] * g[d] + be[d];
  }
}

// ================= launcher =================
extern "C" void kernel_launch(void* const* d_in, const int* in_sizes, int n_in,
                              void* d_out, int out_size, void* d_ws, size_t ws_size,
                              hipStream_t stream) {
  const float* x    = (const float*)d_in[0];
  const float* Cp   = (const float*)d_in[1];
  const float* Wq   = (const float*)d_in[3];
  const float* bq   = (const float*)d_in[4];
  const float* Wk   = (const float*)d_in[5];
  const float* bk   = (const float*)d_in[6];
  const float* Wv   = (const float*)d_in[7];
  const float* bv   = (const float*)d_in[8];
  const float* W1   = (const float*)d_in[9];
  const float* b1   = (const float*)d_in[10];
  const float* W2   = (const float*)d_in[11];
  const float* b2   = (const float*)d_in[12];
  const float* ln1g = (const float*)d_in[13];
  const float* ln1b = (const float*)d_in[14];
  const float* ln2g = (const float*)d_in[15];
  const float* ln2b = (const float*)d_in[16];
  float* out = (float*)d_out;
  (void)in_sizes; (void)n_in; (void)out_size; (void)ws_size;

  char* base = (char*)d_ws;
  size_t off = 0;
  auto carveF = [&](size_t n) { float* p = (float*)(base + off); off = (off + n * 4 + 15) & ~15ULL; return p; };
  auto carveH = [&](size_t n) { _Float16* p = (_Float16*)(base + off); off = (off + n * 2 + 15) & ~15ULL; return p; };

  _Float16* WkT   = carveH((size_t)DIM * DIM);
  _Float16* WvT   = carveH((size_t)DIM * DIM);
  _Float16* W1T   = carveH((size_t)DIM * FF);
  _Float16* W2T   = carveH((size_t)FF * DIM);
  _Float16* K1h   = carveH((size_t)B * K * DIM);
  _Float16* attn2 = carveH((size_t)B * H * K * K);
  _Float16* w2T   = carveH((size_t)B * H * K * HD);
  _Float16* MmT   = carveH((size_t)B * H * K * HD);
  float*    cmax  = carveF((size_t)B * H * K);
  float*    csum  = carveF((size_t)B * H * K);
  float*    wpart = carveF((size_t)NCH * B * H * K * HD);
  float*    rmax  = carveF((size_t)B * H * N);
  float*    rsum  = carveF((size_t)B * H * N);
  _Float16* xs16  = carveH((size_t)B * N * DIM);
  _Float16* Q16   = carveH((size_t)B * N * DIM);
  _Float16* V16   = carveH((size_t)B * N * DIM);
  _Float16* V1t   = carveH((size_t)B * N * DIM);
  _Float16* att3t = carveH((size_t)B * H * N * K);
  _Float16* fin16 = carveH((size_t)B * N * DIM);
  _Float16* hb16  = carveH((size_t)B * N * FF);
  float*    Sc    = carveF((size_t)B * H * N * K);
  float*    att   = carveF((size_t)B * N * DIM);
  float*    fin32 = carveF((size_t)B * N * DIM);

  dim3 tb(32, 8);
  k_tr_f2h<<<dim3(DIM / 32, DIM / 32, 1), tb, 0, stream>>>(Wk, WkT, DIM, DIM);
  k_tr_f2h<<<dim3(DIM / 32, DIM / 32, 1), tb, 0, stream>>>(Wv, WvT, DIM, DIM);
  k_tr_f2h<<<dim3(FF / 32, DIM / 32, 1), tb, 0, stream>>>(W1, W1T, DIM, FF);
  k_tr_f2h<<<dim3(DIM / 32, FF / 32, 1), tb, 0, stream>>>(W2, W2T, FF, DIM);
  k_tr_f2h<<<dim3(N / 32, DIM / 32, B), tb, 0, stream>>>(x, xs16, DIM, N);

  k_proj_k1<<<B * 4 * 16, 32, 0, stream>>>(Cp, Wq, bq, K1h);
  k_attn2<<<B * H, 512, 0, stream>>>(K1h, attn2);
  k_qv<<<(B * (N / 16) * 4) / 8, 256, 0, stream>>>(xs16, WkT, bk, Q16);
  k_qv<<<(B * (N / 16) * 4) / 8, 256, 0, stream>>>(xs16, WvT, bv, V16);
  k_tr_h2h<<<dim3(DIM / 32, N / 32, B), tb, 0, stream>>>(V16, V1t, N, DIM);
  k_scores<<<(B * H * (N / 16)) / 8, 256, 0, stream>>>(Q16, K1h, Sc);
  k_rowstats<<<(B * H * N) / 256, 256, 0, stream>>>(Sc, rmax, rsum);
  k_colstats<<<B * H, 256, 0, stream>>>(Sc, cmax, csum);
  k_attn3t<<<B * H * (N / 256), 256, 0, stream>>>(Sc, cmax, att3t);
  k_w2p<<<B * H * NCH, 128, 0, stream>>>(att3t, V1t, wpart);
  k_w2red<<<(B * H * K * HD) / 256, 256, 0, stream>>>(wpart, csum, w2T);
  k_matM<<<B * H, 512, 0, stream>>>(attn2, w2T, MmT);
  k_att_out<<<(B * H * (N / 16)) / 8, 256, 0, stream>>>(Sc, rmax, rsum, MmT, att);
  k_ln1<<<B * N, 256, 0, stream>>>(att, xs16, ln1g, ln1b, fin32, fin16);
  k_ffn1<<<B * N / 16, 256, 0, stream>>>(fin16, W1T, b1, hb16);
  k_ffn2<<<B * N / 16, 128, 0, stream>>>(hb16, W2T, b2, fin32, ln2g, ln2b, out);
}
